// SparseRNN_67800353735071
// MI455X (gfx1250) — compile-verified, matrix operand order checked by reading, not run on hardware
//
#include <hip/hip_runtime.h>
#include <hip/hip_bf16.h>
#include <cstdint>

// ---------------- problem constants ----------------
constexpr int kH = 8192;   // hidden size (== input size)
constexpr int kB = 128;    // batch
constexpr int kT = 64;     // sequence length
constexpr int KC = 32;     // K per WMMA chunk (v_wmma_f32_16x16x32_f16)
constexpr int NWG = 64;    // output columns (W rows) per workgroup
constexpr int THREADS = 256;  // 8 waves (wave32)

typedef _Float16 v16h __attribute__((ext_vector_type(16)));
typedef _Float16 v8h  __attribute__((ext_vector_type(8)));
typedef float    v8f  __attribute__((ext_vector_type(8)));
typedef unsigned int u32x4 __attribute__((ext_vector_type(4)));
typedef int      i32x4 __attribute__((ext_vector_type(4)));
typedef int      i32x8 __attribute__((ext_vector_type(8)));

#if defined(__gfx1250__) && __has_builtin(__builtin_amdgcn_tensor_load_to_lds) && \
    __has_builtin(__builtin_amdgcn_s_wait_tensorcnt)
#define RNN_TDM 1
#else
#define RNN_TDM 0
#endif

#if RNN_TDM
// 2-D Tensor-Data-Mover load: tile (tile_d1 rows x tile_d0 contiguous fp16
// elements), global row stride stride0 (elements), packed into LDS at lds_off.
__device__ __forceinline__ void tdm_load_2d(unsigned lds_off, const void* gptr,
                                            unsigned tile_d0, unsigned tile_d1,
                                            unsigned tdim0, unsigned tdim1,
                                            unsigned long long stride0) {
  unsigned long long ga = (unsigned long long)(uintptr_t)gptr;
  u32x4 g0;
  g0[0] = 1u;                                   // count=1 (valid user descriptor)
  g0[1] = lds_off;                              // LDS byte address
  g0[2] = (unsigned)ga;                         // global_addr[31:0]
  g0[3] = (unsigned)((ga >> 32) & 0x01FFFFFFull) | 0x80000000u; // addr[56:32]|type=2
  i32x8 g1;
  g1[0] = (int)(1u << 16);                      // data_size=1 -> 2 bytes/element
  g1[1] = (int)((tdim0 & 0xFFFFu) << 16);       // tensor_dim0[15:0] @bit48
  g1[2] = (int)((tdim0 >> 16) | ((tdim1 & 0xFFFFu) << 16));
  g1[3] = (int)((tdim1 >> 16) | (tile_d0 << 16)); // tile_dim0 @bit112
  g1[4] = (int)(tile_d1 & 0xFFFFu);             // tile_dim1 (tile_dim2=0)
  g1[5] = (int)(unsigned)(stride0 & 0xFFFFFFFFull);   // dim0 stride lo
  g1[6] = (int)(unsigned)((stride0 >> 32) & 0xFFFFull);
  g1[7] = 0;
  i32x4 z4 = {0, 0, 0, 0};
#if __clang_major__ >= 23
  i32x8 z8 = {0, 0, 0, 0, 0, 0, 0, 0};
  __builtin_amdgcn_tensor_load_to_lds(g0, g1, z4, z4, z8, 0);
#else
  __builtin_amdgcn_tensor_load_to_lds(g0, g1, z4, z4, 0);
#endif
}
#endif

// Fragment load matching CDNA5 16-bit A/B striping: lane half (lane>>4) picks
// K-groups {halfk*8..+8} and {16+halfk*8..+8} -> two 16B contiguous LDS reads.
__device__ __forceinline__ v16h load_frag(const _Float16* rowp, int halfk) {
  const _Float16* p = rowp + halfk * 8;
  v8h lo = *(const v8h*)p;
  v8h hi = *(const v8h*)(p + 16);
  v16h f;
#pragma unroll
  for (int i = 0; i < 8; ++i) { f[i] = lo[i]; f[8 + i] = hi[i]; }
  return f;
}

// One RNN timestep: D[b][h] = sum_k x[b,k]*Wih[h,k] + sum_k h[b,k]*Whh[h,k];
// then tanh + bias, write out[b][t][h] (f32) and hnext[b][h] (f16).
__global__ __launch_bounds__(THREADS) void rnn_step_kernel(
    const _Float16* __restrict__ xt,     // [kB][kH]
    const _Float16* __restrict__ hcur,   // [kB][kH]
    const _Float16* __restrict__ Wih,    // [kH][kH] row-major
    const _Float16* __restrict__ Whh,    // [kH][kH] row-major
    const float* __restrict__ bias_ih,
    const float* __restrict__ bias_hh,
    float* __restrict__ out,             // [kB][kT][kH]
    _Float16* __restrict__ hnext,        // [kB][kH]
    int tstep) {
  __shared__ __attribute__((aligned(16))) _Float16 Ald[2][kB * KC];   // 2 x 8 KB
  __shared__ __attribute__((aligned(16))) _Float16 Bld[2][NWG * KC];  // 2 x 4 KB

  const int tid = threadIdx.x;
  const int wave = tid >> 5;
  const int lane = tid & 31;
  const int lane16 = lane & 15;
  const int halfk = lane >> 4;
  const int n0 = blockIdx.x * NWG;  // base output column (h)
  const int nt = wave & 3;          // n-tile (16 h columns) owned by this wave
  const int bq = wave >> 2;         // batch half: b-tiles [bq*4, bq*4+4)

  v8f acc[4] = {};
  constexpr int CHUNKS_PER_PHASE = kH / KC;     // 256
  constexpr int NCHUNK = 2 * CHUNKS_PER_PHASE;  // phase0: Wih*x, phase1: Whh*h

#if RNN_TDM
  auto issue_pair = [&](int c) {
    const int buf = c & 1;
    const int phase = c >> 8;
    const int k0 = (c & (CHUNKS_PER_PHASE - 1)) * KC;
    const _Float16* Ab = (phase ? hcur : xt) + k0;
    const _Float16* Bb = (phase ? Whh : Wih) + (size_t)n0 * kH + k0;
    tdm_load_2d((unsigned)(uintptr_t)&Ald[buf][0], Ab, KC, kB,
                (unsigned)kH, (unsigned)kB, (unsigned long long)kH);
    tdm_load_2d((unsigned)(uintptr_t)&Bld[buf][0], Bb, KC, NWG,
                (unsigned)kH, (unsigned)kH, (unsigned long long)kH);
  };
  if (wave == 0) issue_pair(0);
  for (int c = 0; c < NCHUNK; ++c) {
    __syncthreads();  // all waves done with buffer (c+1)&1 before refill
    if (wave == 0) {
      if (c + 1 < NCHUNK) {
        issue_pair(c + 1);
        __builtin_amdgcn_s_wait_tensorcnt(2);  // chunk c's 2 TDM ops retired
      } else {
        __builtin_amdgcn_s_wait_tensorcnt(0);
      }
    }
    __syncthreads();  // chunk c tiles visible in LDS
    const int buf = c & 1;
    v16h bfrag = load_frag(&Bld[buf][(nt * 16 + lane16) * KC], halfk);
#pragma unroll
    for (int bt = 0; bt < 4; ++bt) {
      v16h afrag = load_frag(&Ald[buf][((bq * 4 + bt) * 16 + lane16) * KC], halfk);
      acc[bt] = __builtin_amdgcn_wmma_f32_16x16x32_f16(
          false, afrag, false, bfrag, (short)0, acc[bt], false, false);
    }
  }
#else
  // Fallback: cooperative vector loads (also what the host pass parses).
  for (int c = 0; c < NCHUNK; ++c) {
    const int phase = c >> 8;
    const int k0 = (c & (CHUNKS_PER_PHASE - 1)) * KC;
    const _Float16* Ab = (phase ? hcur : xt) + k0;
    const _Float16* Bb = (phase ? Whh : Wih) + (size_t)n0 * kH + k0;
    for (int i = tid; i < kB * (KC / 8); i += THREADS) {
      int r = i >> 2, s = i & 3;
      *(v8h*)&Ald[0][r * KC + s * 8] = *(const v8h*)(Ab + (size_t)r * kH + s * 8);
    }
    for (int i = tid; i < NWG * (KC / 8); i += THREADS) {
      int r = i >> 2, s = i & 3;
      *(v8h*)&Bld[0][r * KC + s * 8] = *(const v8h*)(Bb + (size_t)r * kH + s * 8);
    }
    __syncthreads();
    v16h bfrag = load_frag(&Bld[0][(nt * 16 + lane16) * KC], halfk);
#pragma unroll
    for (int bt = 0; bt < 4; ++bt) {
      v16h afrag = load_frag(&Ald[0][((bq * 4 + bt) * 16 + lane16) * KC], halfk);
      acc[bt] = __builtin_amdgcn_wmma_f32_16x16x32_f16(
          false, afrag, false, bfrag, (short)0, acc[bt], false, false);
    }
    __syncthreads();
  }
#endif

  // Epilogue. C/D layout: lane16 -> n (h column), vgpr r + 8*halfk -> m (batch).
  const int h = n0 + nt * 16 + lane16;
  const float bsum = bias_ih[h] + bias_hh[h];
#pragma unroll
  for (int bt = 0; bt < 4; ++bt) {
    const int btile = bq * 4 + bt;
#pragma unroll
    for (int r = 0; r < 8; ++r) {
      const int b = btile * 16 + r + 8 * halfk;
      const float v = tanhf(acc[bt][r] + bsum);
      out[(size_t)b * (kT * kH) + (size_t)tstep * kH + h] = v;
      hnext[(size_t)b * kH + h] = (_Float16)v;
    }
  }
}

// ---------------- preprocessing kernels ----------------
constexpr float kFxScale = 1048576.0f;  // 2^20 fixed point for deterministic dup-adds

__global__ void coo_scatter_kernel(const int* __restrict__ idx,
                                   const float* __restrict__ val,
                                   int* __restrict__ Wfx, int nnz) {
  int i = blockIdx.x * blockDim.x + threadIdx.x;
  if (i < nnz) {
    const int r = idx[i];
    const int c = idx[nnz + i];
    const int q = __float2int_rn(val[i] * kFxScale);
    atomicAdd(Wfx + (size_t)r * kH + c, q);
  }
}

__global__ void fx_to_f16_kernel(const int* __restrict__ s,
                                 _Float16* __restrict__ d, size_t n) {
  size_t i = (size_t)blockIdx.x * blockDim.x + threadIdx.x;
  const size_t stride = (size_t)gridDim.x * blockDim.x;
  for (; i < n; i += stride)
    d[i] = (_Float16)((float)s[i] * (1.0f / kFxScale));
}

__global__ void xpose_f16_kernel(const float* __restrict__ x,
                                 _Float16* __restrict__ x16, size_t n) {
  // x16[t][b][h] = (f16) x[b][t][h]  (H-contiguous on both sides)
  size_t i = (size_t)blockIdx.x * blockDim.x + threadIdx.x;
  const size_t stride = (size_t)gridDim.x * blockDim.x;
  for (; i < n; i += stride) {
    const size_t h = i % kH;
    const size_t tb = i / kH;
    const size_t b = tb % kB;
    const size_t t = tb / kB;
    x16[i] = (_Float16)x[(b * kT + t) * kH + h];
  }
}

// ---------------- host launch ----------------
extern "C" void kernel_launch(void* const* d_in, const int* in_sizes, int n_in,
                              void* d_out, int out_size, void* d_ws, size_t ws_size,
                              hipStream_t stream) {
  const float* x      = (const float*)d_in[0];
  const int*   ih_idx = (const int*)d_in[1];
  const float* ih_val = (const float*)d_in[2];
  const int*   hh_idx = (const int*)d_in[3];
  const float* hh_val = (const float*)d_in[4];
  const float* b_ih   = (const float*)d_in[5];
  const float* b_hh   = (const float*)d_in[6];
  float* out = (float*)d_out;
  const int nnz_ih = in_sizes[2];
  const int nnz_hh = in_sizes[4];
  (void)n_in; (void)out_size; (void)ws_size;  // needs ~645 MB scratch

  char* ws = (char*)d_ws;
  size_t off = 0;
  _Float16* Wih  = (_Float16*)(ws + off); off += (size_t)kH * kH * sizeof(_Float16);
  _Float16* Whh  = (_Float16*)(ws + off); off += (size_t)kH * kH * sizeof(_Float16);
  _Float16* x16  = (_Float16*)(ws + off); off += (size_t)kT * kB * kH * sizeof(_Float16);
  _Float16* h16a = (_Float16*)(ws + off); off += (size_t)kB * kH * sizeof(_Float16);
  _Float16* h16b = (_Float16*)(ws + off); off += (size_t)kB * kH * sizeof(_Float16);
  int*      Wfx  = (int*)(ws + off);      off += (size_t)kH * kH * sizeof(int);

  const size_t wn = (size_t)kH * kH;
  // densify ih -> fp16 (fixed-point atomics make duplicate nnz deterministic)
  hipMemsetAsync(Wfx, 0, wn * sizeof(int), stream);
  coo_scatter_kernel<<<(nnz_ih + 255) / 256, 256, 0, stream>>>(ih_idx, ih_val, Wfx, nnz_ih);
  fx_to_f16_kernel<<<8192, 256, 0, stream>>>(Wfx, Wih, wn);
  // densify hh -> fp16
  hipMemsetAsync(Wfx, 0, wn * sizeof(int), stream);
  coo_scatter_kernel<<<(nnz_hh + 255) / 256, 256, 0, stream>>>(hh_idx, hh_val, Wfx, nnz_hh);
  fx_to_f16_kernel<<<8192, 256, 0, stream>>>(Wfx, Whh, wn);
  // x -> fp16, time-major [T][B][H]
  xpose_f16_kernel<<<8192, 256, 0, stream>>>(x, x16, (size_t)kT * kB * kH);
  // h0 = 0
  hipMemsetAsync(h16a, 0, (size_t)kB * kH * sizeof(_Float16), stream);

  _Float16* hc = h16a;
  _Float16* hn = h16b;
  for (int t = 0; t < kT; ++t) {
    rnn_step_kernel<<<kH / NWG, THREADS, 0, stream>>>(
        x16 + (size_t)t * kB * kH, hc, Wih, Whh, b_ih, b_hh, out, hn, t);
    _Float16* tmp = hc; hc = hn; hn = tmp;
  }
}
